// SetAbstractionMSG_74818330296615
// MI455X (gfx1250) — matrix operand order repaired and checked by reading, not compiled, not run on hardware
//
#include <hip/hip_runtime.h>
#include <hip/hip_bf16.h>

typedef __attribute__((ext_vector_type(16))) _Float16 v16h;
typedef __attribute__((ext_vector_type(8)))  _Float16 v8h;
typedef __attribute__((ext_vector_type(8)))  float    v8f;

#define NPTS 4096
#define NQ   1024
#define NB   8
#define DF   32
#define COUT_TOTAL 320
#define XSTRIDE 136   // halves; 272B rows -> bank-spread, 16B aligned
#define WSTRIDE 104   // halves; 208B rows -> 16B aligned

// ---------------------------------------------------------------------------
// Farthest point sampling: 1 block per batch, 256 threads, 16 pts/thread.
// ---------------------------------------------------------------------------
__global__ __launch_bounds__(256)
void fps_kernel(const float* __restrict__ xyz, int* __restrict__ fpsIdx,
                float* __restrict__ nxyz, float* __restrict__ out) {
  const int b = blockIdx.x, tid = threadIdx.x;
  const float* base = xyz + (size_t)b * NPTS * 3;
  float px[16], py[16], pz[16], dd[16];
#pragma unroll
  for (int j = 0; j < 16; ++j) {
    int i = j * 256 + tid;
    px[j] = base[i * 3 + 0];
    py[j] = base[i * 3 + 1];
    pz[j] = base[i * 3 + 2];
    dd[j] = 1e10f;
  }
  __shared__ float sv[256];
  __shared__ int   si[256];
  __shared__ float c0, c1, c2;
  if (tid == 0) {
    c0 = base[0]; c1 = base[1]; c2 = base[2];
    fpsIdx[b * NQ] = 0;
    int o = (b * NQ) * 3;
    nxyz[o] = base[0]; nxyz[o + 1] = base[1]; nxyz[o + 2] = base[2];
    out[o]  = base[0]; out[o + 1]  = base[1]; out[o + 2]  = base[2];
  }
  __syncthreads();
  for (int it = 1; it < NQ; ++it) {
    float cx = c0, cy = c1, cz = c2;
    float bv = -1.0f; int bi = 0x7fffffff;
#pragma unroll
    for (int j = 0; j < 16; ++j) {
      float dx = px[j] - cx, dy = py[j] - cy, dz = pz[j] - cz;
      float d2 = dx * dx + dy * dy + dz * dz;
      dd[j] = fminf(dd[j], d2);
      if (dd[j] > bv) { bv = dd[j]; bi = j * 256 + tid; }
    }
    sv[tid] = bv; si[tid] = bi;
    __syncthreads();
    for (int s = 128; s > 0; s >>= 1) {
      if (tid < s) {
        float v2 = sv[tid + s]; int i2 = si[tid + s];
        if (v2 > sv[tid] || (v2 == sv[tid] && i2 < si[tid])) { sv[tid] = v2; si[tid] = i2; }
      }
      __syncthreads();
    }
    if (tid == 0) {
      int cur = si[0];
      fpsIdx[b * NQ + it] = cur;
      float x0 = base[cur * 3], x1 = base[cur * 3 + 1], x2 = base[cur * 3 + 2];
      c0 = x0; c1 = x1; c2 = x2;
      int o = (b * NQ + it) * 3;
      nxyz[o] = x0; nxyz[o + 1] = x1; nxyz[o + 2] = x2;
      out[o]  = x0; out[o + 1]  = x1; out[o + 2]  = x2;
    }
    __syncthreads();
  }
}

// ---------------------------------------------------------------------------
// Ball query: block caches batch xyz in LDS; 1 thread per query, ascending
// scan keeps first NS in-radius indices (== top-k of masked indices).
// ---------------------------------------------------------------------------
template <int NS>
__global__ __launch_bounds__(256)
void ballq_kernel(const float* __restrict__ xyz, const float* __restrict__ nxyz,
                  int* __restrict__ gidx, float r2) {
  __shared__ float sx[NPTS], sy[NPTS], sz[NPTS];
  const int b = blockIdx.x >> 2;
  const int qblk = blockIdx.x & 3;
  const int tid = threadIdx.x;
  const float* base = xyz + (size_t)b * NPTS * 3;
  for (int i = tid; i < NPTS; i += 256) {
    sx[i] = base[i * 3 + 0];
    sy[i] = base[i * 3 + 1];
    sz[i] = base[i * 3 + 2];
  }
  __syncthreads();
  const int q = qblk * 256 + tid;
  const float cx = nxyz[(b * NQ + q) * 3 + 0];
  const float cy = nxyz[(b * NQ + q) * 3 + 1];
  const float cz = nxyz[(b * NQ + q) * 3 + 2];
  int* g = gidx + ((size_t)(b * NQ + q)) * NS;
  int cnt = 0, first = NPTS - 1;
  for (int i = 0; i < NPTS; ++i) {
    float dx = sx[i] - cx, dy = sy[i] - cy, dz = sz[i] - cz;
    float d2 = dx * dx + dy * dy + dz * dz;
    if (d2 <= r2) {
      if (cnt == 0) first = i;
      g[cnt++] = i;
      if (cnt == NS) break;
    }
  }
  for (int j = cnt; j < NS; ++j) g[j] = first;
}

// ---------------------------------------------------------------------------
// One dense layer via WMMA f32_16x16x32_f16, fragments per CDNA5 ISA layout.
// Each of 4 waves owns 16 rows. A fragments are loop-invariant over the
// N-tile loop, so hoist them into registers once per layer (<= 3 x v16h);
// only B fragments stream from LDS inside the N x K loop.
// ---------------------------------------------------------------------------
template <int KPAD, int COUT>
__device__ inline void mlp_layer(const _Float16* __restrict__ in,
                                 _Float16* __restrict__ outb,
                                 _Float16* __restrict__ Wt,
                                 float* __restrict__ sBias,
                                 const float* __restrict__ Wg,
                                 const float* __restrict__ Bg,
                                 int Kreal, int tid) {
  __syncthreads();  // prior layer done with Wt / X buffers
  // Stage W transposed (Wt[col][k]) with zero K-padding, plus bias.
  for (int i = tid; i < COUT * KPAD; i += 128) {
    int col = i / KPAD, k = i % KPAD;
    Wt[col * WSTRIDE + k] = (k < Kreal) ? (_Float16)Wg[k * COUT + col] : (_Float16)0.0f;
  }
  if (tid < COUT) sBias[tid] = Bg[tid];
  __syncthreads();

  const int lane  = tid & 31;
  const int mbase = (tid >> 5) * 16;
  const int rowA  = mbase + (lane & 15);
  const int kselA = (lane & 16) ? 8 : 0;   // A: lanes 0-15 K{0-7,16-23}, 16-31 K{8-15,24-31}
  const int colB  = lane & 15;
  const int kselB = (lane & 16) ? 8 : 0;   // B: lanes 0-15 K{0-7,16-23}, 16-31 K{8-15,24-31}
  const int rsel  = (lane & 16) ? 8 : 0;   // D: lanes 0-15 M 0-7,  16-31 M 8-15

  constexpr int NK = KPAD / 32;
  v16h Afrag[NK];
#pragma unroll
  for (int kc = 0; kc < NK; ++kc) {
    v8h alo = *(const v8h*)&in[rowA * XSTRIDE + kc * 32 + kselA];
    v8h ahi = *(const v8h*)&in[rowA * XSTRIDE + kc * 32 + kselA + 16];
#pragma unroll
    for (int i = 0; i < 8; ++i) { Afrag[kc][i] = alo[i]; Afrag[kc][8 + i] = ahi[i]; }
  }

#pragma unroll
  for (int n0 = 0; n0 < COUT; n0 += 16) {
    v8f acc = {};
#pragma unroll
    for (int kc = 0; kc < NK; ++kc) {
      v8h blo = *(const v8h*)&Wt[colB * WSTRIDE + kc * 32 + kselB];
      v8h bhi = *(const v8h*)&Wt[colB * WSTRIDE + kc * 32 + kselB + 16];
      v16h Bv;
#pragma unroll
      for (int i = 0; i < 8; ++i) { Bv[i] = blo[i]; Bv[8 + i] = bhi[i]; }
      acc = __builtin_amdgcn_wmma_f32_16x16x32_f16(false, Afrag[kc], false, Bv,
                                                   (short)0, acc, false, false);
    }
    const int col = n0 + colB;
    const float bia = sBias[col];
#pragma unroll
    for (int v = 0; v < 8; ++v) {
      float val = acc[v] + bia;
      val = val > 0.0f ? val : 0.0f;  // ReLU
      outb[(mbase + rsel + v) * XSTRIDE + col] = (_Float16)val;
    }
  }
}

// ---------------------------------------------------------------------------
// Grouping + 3-layer MLP + maxpool. 64-row M-tile per block (64/NS queries).
// ---------------------------------------------------------------------------
template <int NS, int C1, int C2, int C3, int OFF>
__global__ __launch_bounds__(128)
void mlp_kernel(const float* __restrict__ xyz, const float* __restrict__ feat,
                const float* __restrict__ nxyz, const int* __restrict__ gidx,
                const float* __restrict__ W1, const float* __restrict__ B1,
                const float* __restrict__ W2, const float* __restrict__ B2,
                const float* __restrict__ W3, const float* __restrict__ B3,
                float* __restrict__ out) {
  __shared__ __align__(16) _Float16 Xa[64 * XSTRIDE];
  __shared__ __align__(16) _Float16 Xb[64 * XSTRIDE];
  __shared__ __align__(16) _Float16 Wt[128 * WSTRIDE];
  __shared__ float sBias[128];

  constexpr int QPB = 64 / NS;
  const int blk = blockIdx.x;
  const int b   = blk / (NQ / QPB);
  const int q0  = (blk % (NQ / QPB)) * QPB;
  const int tid = threadIdx.x;

  // Stage X input tile (64 rows x 64 cols, cols 0-31 feat, 32-34 rel-xyz, rest 0)
  {
    const int r = tid & 63, part = tid >> 6;
    const int q = q0 + r / NS, s = r % NS;
    const int idx = gidx[((size_t)(b * NQ + q)) * NS + s];
    _Float16* row = &Xa[r * XSTRIDE];
    if (part == 0) {
      const float* fp = feat + ((size_t)b * NPTS + idx) * DF;
#pragma unroll
      for (int c = 0; c < DF; ++c) row[c] = (_Float16)fp[c];
    } else {
      const float* pp = xyz + ((size_t)b * NPTS + idx) * 3;
      const float* np = nxyz + ((size_t)b * NQ + q) * 3;
      row[32] = (_Float16)(pp[0] - np[0]);
      row[33] = (_Float16)(pp[1] - np[1]);
      row[34] = (_Float16)(pp[2] - np[2]);
#pragma unroll
      for (int c = 35; c < 64; ++c) row[c] = (_Float16)0.0f;
    }
  }

  mlp_layer<64, C1>(Xa, Xb, Wt, sBias, W1, B1, DF + 3, tid);
  mlp_layer<C1, C2>(Xb, Xa, Wt, sBias, W2, B2, C1, tid);
  mlp_layer<C2, C3>(Xa, Xb, Wt, sBias, W3, B3, C2, tid);
  __syncthreads();

  // Maxpool over NS samples per query, write into concat slice [OFF, OFF+C3)
  if (tid < C3) {
    for (int q = 0; q < QPB; ++q) {
      float m = -1e30f;
#pragma unroll 4
      for (int s = 0; s < NS; ++s)
        m = fmaxf(m, (float)Xb[(q * NS + s) * XSTRIDE + tid]);
      out[(size_t)NB * NQ * 3 + ((size_t)(b * NQ + q0 + q)) * COUT_TOTAL + OFF + tid] = m;
    }
  }
}

// ---------------------------------------------------------------------------
extern "C" void kernel_launch(void* const* d_in, const int* in_sizes, int n_in,
                              void* d_out, int out_size, void* d_ws, size_t ws_size,
                              hipStream_t stream) {
  (void)in_sizes; (void)n_in; (void)out_size; (void)ws_size;
  const float* xyz  = (const float*)d_in[0];
  const float* feat = (const float*)d_in[1];
  const float* W[3][3]; const float* Bp[3][3];
  int p = 2;
  for (int s = 0; s < 3; ++s)
    for (int l = 0; l < 3; ++l) { W[s][l] = (const float*)d_in[p++]; Bp[s][l] = (const float*)d_in[p++]; }
  float* out = (float*)d_out;

  char* ws = (char*)d_ws;
  int*   fps  = (int*)ws;                         // 8*1024 ints
  float* nxyz = (float*)(ws + 32768);             // 8*1024*3 floats
  int*   g0   = (int*)(ws + 32768 + 98304);       // 8*1024*16
  int*   g1   = g0 + NB * NQ * 16;                // 8*1024*32
  int*   g2   = g1 + NB * NQ * 32;                // 8*1024*64

  fps_kernel<<<NB, 256, 0, stream>>>(xyz, fps, nxyz, out);

  ballq_kernel<16><<<NB * 4, 256, 0, stream>>>(xyz, nxyz, g0, 0.01f);
  ballq_kernel<32><<<NB * 4, 256, 0, stream>>>(xyz, nxyz, g1, 0.04f);
  ballq_kernel<64><<<NB * 4, 256, 0, stream>>>(xyz, nxyz, g2, 0.16f);

  mlp_kernel<16, 32, 32, 64, 0><<<NB * NQ / 4, 128, 0, stream>>>(
      xyz, feat, nxyz, g0, W[0][0], Bp[0][0], W[0][1], Bp[0][1], W[0][2], Bp[0][2], out);
  mlp_kernel<32, 64, 64, 128, 64><<<NB * NQ / 2, 128, 0, stream>>>(
      xyz, feat, nxyz, g1, W[1][0], Bp[1][0], W[1][1], Bp[1][1], W[1][2], Bp[1][2], out);
  mlp_kernel<64, 64, 96, 128, 192><<<NB * NQ, 128, 0, stream>>>(
      xyz, feat, nxyz, g2, W[2][0], Bp[2][0], W[2][1], Bp[2][1], W[2][2], Bp[2][2], out);
}
